// QGNN_75445395522274
// MI455X (gfx1250) — matrix operand order repaired and checked by reading, not compiled, not run on hardware
//
#include <hip/hip_runtime.h>

// ---------------------------------------------------------------------------
// QGNN on MI455X (gfx1250): 5 fused layers of edge-conditioned graph conv.
// Matrix math runs on v_wmma_f32_16x16x32_bf16 (wave32 WMMA).
// ---------------------------------------------------------------------------

typedef unsigned short u16;
typedef __attribute__((ext_vector_type(16))) __bf16 v16bf;
typedef __attribute__((ext_vector_type(8)))  float  v8f;

union Frag {
    v16bf v;
    uint4 q[2];
    u16   u[16];
};

__device__ __forceinline__ u16 f2bf(float f) {
    union { float f; unsigned u; } x;
    x.f = f;
    unsigned u = x.u;
    // round-to-nearest-even f32 -> bf16
    return (u16)((u + 0x7FFFu + ((u >> 16) & 1u)) >> 16);
}

// --------------------------- utility kernels -------------------------------

__global__ void zero_f32(float* __restrict__ p, int n) {
    int i = blockIdx.x * blockDim.x + threadIdx.x;
    if (i < n) p[i] = 0.0f;
}

__global__ void count_deg(const int* __restrict__ dst, float* __restrict__ cnt, int E) {
    int i = blockIdx.x * blockDim.x + threadIdx.x;
    if (i < E) atomicAdd(&cnt[dst[i]], 1.0f);
}

__global__ void invert_cnt(float* __restrict__ cnt, int N) {
    int i = blockIdx.x * blockDim.x + threadIdx.x;
    if (i < N) cnt[i] = 1.0f / fmaxf(cnt[i], 1.0f);
}

// h0[n,k] = bf16(emb[gate_type[n], k]),  k in [0,64)
__global__ void embed_kernel(const int* __restrict__ gt, const float* __restrict__ emb,
                             u16* __restrict__ h, int N) {
    int i = blockIdx.x * blockDim.x + threadIdx.x;
    if (i < N * 64) {
        int n = i >> 6, k = i & 63;
        h[i] = f2bf(emb[gt[n] * 64 + k]);
    }
}

// Pack a row-major [rows, Kin] f32 weight into WMMA-B bf16 fragment order:
// fragment (nt, kt): lane l holds column n = nt*16 + (l&15), 16 K values
// K = kt*32 + (e>>3)*16 + (l>>4)*8 + (e&7).  Zero-padded for K >= Kin.
__global__ void pack_w(const float* __restrict__ W, u16* __restrict__ out,
                       int ksteps, int Kin, int total) {
    int idx = blockIdx.x * blockDim.x + threadIdx.x;
    if (idx >= total) return;
    int e    = idx & 15;
    int lane = (idx >> 4) & 31;
    int t    = idx >> 9;
    int kt   = t % ksteps;
    int nt   = t / ksteps;
    int n = nt * 16 + (lane & 15);
    int K = kt * 32 + (e >> 3) * 16 + (lane >> 4) * 8 + (e & 7);
    float val = (K < Kin) ? W[(size_t)n * Kin + K] : 0.0f;
    out[idx] = f2bf(val);
}

// ------------------------- edge kernel (layer front) -----------------------
// One wave per 16-edge tile.  tmp = leaky_relu([h[src]||w_edge] @ W1^T),
// then atomic f32 scatter-add into s[dst] (segment sum; mean applied later).
// Requires E % 16 == 0 (holds: 400000).
__global__ void __launch_bounds__(256)
edge_qconv(const u16* __restrict__ h_bf, const int* __restrict__ src,
           const int* __restrict__ dst, const float* __restrict__ src_idx,
           const float* __restrict__ dst_idx, const float* __restrict__ rev,
           const u16* __restrict__ W1bf, float* __restrict__ s,
           int d, int ksteps, int ntilesE)
{
    int wave = (int)((blockIdx.x * blockDim.x + threadIdx.x) >> 5);
    if (wave >= ntilesE) return;           // wave-uniform exit, EXEC stays all-ones
    int lane = threadIdx.x & 31;
    int hb   = lane >> 4;                  // half-wave select
    int r    = lane & 15;                  // A row / C column
    int e0   = wave * 16;

    int erow = e0 + r;
    int srow = src[erow];
    const u16* hrow = h_bf + (size_t)srow * d;
    float ef0 = src_idx[erow], ef1 = dst_idx[erow], ef2 = rev[erow];

    int drow[8];
#pragma unroll
    for (int v = 0; v < 8; ++v) drow[v] = dst[e0 + hb * 8 + v];

    v8f acc[8];
#pragma unroll
    for (int nt = 0; nt < 8; ++nt) { v8f z = {}; acc[nt] = z; }

    const uint4* wb = (const uint4*)W1bf;
    for (int kt = 0; kt < ksteps; ++kt) {
        // Build A fragment: two contiguous 8-elem K runs per lane.
        Frag a;
#pragma unroll
        for (int j = 0; j < 2; ++j) {
            int kb = kt * 32 + j * 16 + hb * 8;   // multiple of 8
            if (kb + 8 <= d) {
                a.q[j] = *(const uint4*)(hrow + kb);   // gathered node features
            } else {                                   // padded tail: edge scalars
#pragma unroll
                for (int t = 0; t < 8; ++t) {
                    int K = kb + t;
                    float val = 0.0f;
                    if (K == d)          val = ef0;
                    else if (K == d + 1) val = ef1;
                    else if (K == d + 2) val = ef2;
                    a.u[j * 8 + t] = f2bf(val);
                }
            }
        }
#pragma unroll
        for (int nt = 0; nt < 8; ++nt) {
            Frag b;
            size_t off = ((size_t)(nt * ksteps + kt) * 32 + lane) * 2;
            b.q[0] = wb[off];
            b.q[1] = wb[off + 1];
            acc[nt] = __builtin_amdgcn_wmma_f32_16x16x32_bf16(
                false, a.v, false, b.v, (short)0, acc[nt], false, false);
        }
    }

    // leaky_relu + segment-sum scatter (f32 atomics into L2)
#pragma unroll
    for (int nt = 0; nt < 8; ++nt) {
#pragma unroll
        for (int v = 0; v < 8; ++v) {
            float x = acc[nt][v];
            x = x > 0.0f ? x : 0.01f * x;
            atomicAdd(&s[(size_t)drow[v] * 128 + nt * 16 + r], x);
        }
    }
}

// ------------------------- node kernel (layer back) ------------------------
// out = [h || s*inv] @ W2^T + b2, optional ReLU.  Fuses mean + concat + GEMM.
// Requires N % 16 == 0 (holds: 100000).
template <int NT>
__global__ void __launch_bounds__(256)
node_linear2(const u16* __restrict__ h_bf, const float* __restrict__ s,
             const float* __restrict__ inv, const u16* __restrict__ W2bf,
             const float* __restrict__ b2, u16* __restrict__ h_out,
             float* __restrict__ f_out, int d, int ksteps, int fo,
             int ntilesN, int do_relu)
{
    int wave = (int)((blockIdx.x * blockDim.x + threadIdx.x) >> 5);
    if (wave >= ntilesN) return;
    int lane = threadIdx.x & 31;
    int hb   = lane >> 4;
    int r    = lane & 15;
    int n0   = wave * 16;

    int nrow = n0 + r;
    const u16*   hrow = h_bf + (size_t)nrow * d;
    const float* srow = s + (size_t)nrow * 128;
    float iv = inv[nrow];

    v8f acc[NT];
#pragma unroll
    for (int nt = 0; nt < NT; ++nt) { v8f z = {}; acc[nt] = z; }

    const uint4* wb = (const uint4*)W2bf;
    for (int kt = 0; kt < ksteps; ++kt) {
        Frag a;
#pragma unroll
        for (int j = 0; j < 2; ++j) {
            int kb = kt * 32 + j * 16 + hb * 8;
            if (kb + 8 <= d) {
                a.q[j] = *(const uint4*)(hrow + kb);          // h part (bf16)
            } else {                                          // h_N part = s * inv
                const float4 f1 = *(const float4*)(srow + (kb - d));
                const float4 f2 = *(const float4*)(srow + (kb - d) + 4);
                a.u[j * 8 + 0] = f2bf(f1.x * iv);
                a.u[j * 8 + 1] = f2bf(f1.y * iv);
                a.u[j * 8 + 2] = f2bf(f1.z * iv);
                a.u[j * 8 + 3] = f2bf(f1.w * iv);
                a.u[j * 8 + 4] = f2bf(f2.x * iv);
                a.u[j * 8 + 5] = f2bf(f2.y * iv);
                a.u[j * 8 + 6] = f2bf(f2.z * iv);
                a.u[j * 8 + 7] = f2bf(f2.w * iv);
            }
        }
#pragma unroll
        for (int nt = 0; nt < NT; ++nt) {
            Frag b;
            size_t off = ((size_t)(nt * ksteps + kt) * 32 + lane) * 2;
            b.q[0] = wb[off];
            b.q[1] = wb[off + 1];
            acc[nt] = __builtin_amdgcn_wmma_f32_16x16x32_bf16(
                false, a.v, false, b.v, (short)0, acc[nt], false, false);
        }
    }

    // epilogue: bias, relu, write bf16 activation and/or f32 final output
#pragma unroll
    for (int nt = 0; nt < NT; ++nt) {
        int c = nt * 16 + r;
        float bias = b2[c];
#pragma unroll
        for (int v = 0; v < 8; ++v) {
            int node = n0 + hb * 8 + v;
            float x = acc[nt][v] + bias;
            if (do_relu) x = fmaxf(x, 0.0f);
            if (h_out) h_out[(size_t)node * fo + c] = f2bf(x);
            if (f_out) f_out[(size_t)node * fo + c] = x;
        }
    }
}

// ------------------------------- launcher ----------------------------------

extern "C" void kernel_launch(void* const* d_in, const int* in_sizes, int n_in,
                              void* d_out, int out_size, void* d_ws, size_t ws_size,
                              hipStream_t stream) {
    (void)in_sizes; (void)n_in; (void)out_size; (void)ws_size;

    const int*   gate_type = (const int*)d_in[0];
    const int*   src       = (const int*)d_in[1];
    const int*   dst       = (const int*)d_in[2];
    const float* src_idx   = (const float*)d_in[3];
    const float* dst_idx   = (const float*)d_in[4];
    const float* rev       = (const float*)d_in[5];
    const float* emb       = (const float*)d_in[6];
    const float *w1[5], *w2[5], *b2[5];
    for (int i = 0; i < 5; ++i) {
        w1[i] = (const float*)d_in[7 + 3 * i];
        w2[i] = (const float*)d_in[8 + 3 * i];
        b2[i] = (const float*)d_in[9 + 3 * i];
    }

    constexpr int N = 100000, E = 400000;
    const int fi[5] = {64, 128, 128, 128, 128};
    const int fo[5] = {128, 128, 128, 128, 64};
    int ks1[5], ks2[5];
    for (int i = 0; i < 5; ++i) {
        ks1[i] = (fi[i] + 3 + 31) / 32;   // padded K for linear1
        ks2[i] = (fi[i] + 128) / 32;      // exact K for linear2
    }

    // carve workspace (256B aligned)
    char* ws = (char*)d_ws;
    size_t off = 0;
    auto carve = [&](size_t bytes) -> void* {
        void* p = ws + off;
        off = (off + bytes + 255) & ~(size_t)255;
        return p;
    };
    u16*   hA  = (u16*)carve((size_t)N * 128 * sizeof(u16));
    u16*   hB  = (u16*)carve((size_t)N * 128 * sizeof(u16));
    float* s   = (float*)carve((size_t)N * 128 * sizeof(float));
    float* inv = (float*)carve((size_t)N * sizeof(float));
    u16* W1p[5]; u16* W2p[5];
    for (int i = 0; i < 5; ++i) {
        W1p[i] = (u16*)carve((size_t)8 * ks1[i] * 512 * sizeof(u16));
        W2p[i] = (u16*)carve((size_t)(fo[i] / 16) * ks2[i] * 512 * sizeof(u16));
    }

    // --- one-time prep (runs each launch; deterministic) ---
    zero_f32<<<(N + 255) / 256, 256, 0, stream>>>(inv, N);
    count_deg<<<(E + 255) / 256, 256, 0, stream>>>(dst, inv, E);
    invert_cnt<<<(N + 255) / 256, 256, 0, stream>>>(inv, N);
    embed_kernel<<<(N * 64 + 255) / 256, 256, 0, stream>>>(gate_type, emb, hA, N);
    for (int i = 0; i < 5; ++i) {
        int t1 = 8 * ks1[i] * 512;
        pack_w<<<(t1 + 255) / 256, 256, 0, stream>>>(w1[i], W1p[i], ks1[i], fi[i] + 3, t1);
        int t2 = (fo[i] / 16) * ks2[i] * 512;
        pack_w<<<(t2 + 255) / 256, 256, 0, stream>>>(w2[i], W2p[i], ks2[i], fi[i] + 128, t2);
    }

    // --- 5 layers ---
    u16* hin = hA; u16* hout = hB;
    const int etiles  = E / 16;   // 25000 (exact)
    const int ntilesN = N / 16;   // 6250 (exact)
    for (int i = 0; i < 5; ++i) {
        zero_f32<<<(N * 128 + 255) / 256, 256, 0, stream>>>(s, N * 128);
        edge_qconv<<<(etiles + 7) / 8, 256, 0, stream>>>(
            hin, src, dst, src_idx, dst_idx, rev, W1p[i], s, fi[i], ks1[i], etiles);
        bool last = (i == 4);
        if (fo[i] == 128) {
            node_linear2<8><<<(ntilesN + 7) / 8, 256, 0, stream>>>(
                hin, s, inv, W2p[i], b2[i],
                last ? nullptr : hout, last ? (float*)d_out : nullptr,
                fi[i], ks2[i], fo[i], ntilesN, last ? 0 : 1);
        } else {
            node_linear2<4><<<(ntilesN + 7) / 8, 256, 0, stream>>>(
                hin, s, inv, W2p[i], b2[i],
                last ? nullptr : hout, last ? (float*)d_out : nullptr,
                fi[i], ks2[i], fo[i], ntilesN, last ? 0 : 1);
        }
        u16* t = hin; hin = hout; hout = t;
    }
}
